// RnnBlock_7791070675588
// MI455X (gfx1250) — matrix operand order repaired and checked by reading, not compiled, not run on hardware
//
#include <hip/hip_runtime.h>

typedef __attribute__((ext_vector_type(2)))  float  v2f;
typedef __attribute__((ext_vector_type(8)))  float  v8f;
typedef __attribute__((ext_vector_type(8)))  __bf16 v8bf;
typedef __attribute__((ext_vector_type(16))) __bf16 v16bf;

#define B_TOT 1024
#define C_IN  64
#define T_LEN 1024
#define H_DIM 32
#define G_DIM 96
#define TT    8                    // time steps per x-chunk
#define BT    16                   // batch rows per workgroup
#define HBS   40                   // hb row stride: 80 B -> aligned b128 A loads
#define XROW  (C_IN + 4)           // 68 floats: conflict-free A reads
#define XPLANE (BT * XROW + 8)     // 1096 dwords: +8 breaks tt-plane bank aliasing
#define NCHUNK (T_LEN / TT)

__device__ __forceinline__ float fast_sigmoid(float x) {
    return __builtin_amdgcn_rcpf(1.f + __expf(-x));
}
__device__ __forceinline__ float fast_tanh(float x) {
    float e = __expf(2.f * x);
    return 1.f - 2.f * __builtin_amdgcn_rcpf(e + 1.f);
}

__launch_bounds__(192, 1)
__global__ void gru_fused_kernel(const float* __restrict__ x,
                                 const float* __restrict__ W_ih,
                                 const float* __restrict__ W_hh,
                                 const float* __restrict__ b_ih,
                                 const float* __restrict__ b_hh,
                                 float* __restrict__ out)
{
    __shared__ float  xc[2][TT * XPLANE];             // double-buffered x chunks
    __shared__ float  hf[BT][H_DIM + 1];              // f32 carry-path h
    __shared__ __align__(16) __bf16 hb[BT][HBS];      // bf16 h (WMMA A operand)
    __shared__ float  rbuf[BT][H_DIM + 1];
    __shared__ float  zbuf[BT][H_DIM + 1];

    const int tid  = threadIdx.x;
    const int wv   = tid >> 5;      // wave id 0..5 == gate-column tile
    const int lane = tid & 31;      // wave32
    const int half = lane >> 4;     // 0: lanes 0-15, 1: lanes 16-31
    const int nl   = lane & 15;     // N (or M) index within 16-wide tile
    const int b0   = blockIdx.x * BT;

    // ---- preload B operands (kept in VGPRs for the whole kernel) ----
    v2f wih[16];
#pragma unroll
    for (int ks = 0; ks < 16; ++ks) {
        const float* p = W_ih + (size_t)(16 * wv + nl) * C_IN + (4 * ks + 2 * half);
        wih[ks].x = p[0];
        wih[ks].y = p[1];
    }
    v16bf whh;
#pragma unroll
    for (int v = 0; v < 8; ++v) {
        const float* p = W_hh + (size_t)(16 * wv + nl) * H_DIM + (2 * v + 16 * half);
        whh[2 * v]     = (__bf16)p[0];
        whh[2 * v + 1] = (__bf16)p[1];
    }
    const float bih     = b_ih[16 * wv + nl];
    const float bhh     = b_hh[16 * wv + nl];
    const float bias_rz = bih + bhh;

    // zero hidden state
    for (int i = tid; i < BT * HBS; i += blockDim.x) {
        hb[i / HBS][i % HBS] = (__bf16)0.f;
        if (i < BT * H_DIM) hf[i >> 5][i & 31] = 0.f;
    }

    // ---- async prefetch of chunk 0 into buffer 0 ----
    // element e enumerated (b,c,tt): per-instr lanes cover 4 (b,c) x 8 tt,
    // global side = 4 x 32B segments, LDS side bank-spread by XPLANE pad.
    {
        const float* xg = x + (size_t)b0 * (C_IN * T_LEN);
        for (int e = wv * 32 + lane; e < BT * C_IN * TT; e += 192) {
            int tt = e & 7, c = (e >> 3) & 63, b = e >> 9;
            const float* gp = xg + (size_t)b * (C_IN * T_LEN) + (size_t)c * T_LEN + tt;
            unsigned lds = (unsigned)(size_t)(&xc[0][tt * XPLANE + b * XROW + c]);
            asm volatile("global_load_async_to_lds_b32 %0, %1, off"
                         :: "v"(lds), "v"(gp) : "memory");
        }
    }
    asm volatile("s_wait_asynccnt 0x0" ::: "memory");
    __syncthreads();

    const v8f vzero = {};
    int cur = 0;

    for (int k = 0; k < NCHUNK; ++k) {
        // ---- async prefetch of next chunk into the other buffer ----
        if (k + 1 < NCHUNK) {
            const int t0n = (k + 1) * TT;
            const float* xg = x + (size_t)b0 * (C_IN * T_LEN);
            float* dst = &xc[cur ^ 1][0];
            for (int e = wv * 32 + lane; e < BT * C_IN * TT; e += 192) {
                int tt = e & 7, c = (e >> 3) & 63, b = e >> 9;
                const float* gp = xg + (size_t)b * (C_IN * T_LEN)
                                     + (size_t)c * T_LEN + (t0n + tt);
                unsigned lds = (unsigned)(size_t)(dst + tt * XPLANE + b * XROW + c);
                asm volatile("global_load_async_to_lds_b32 %0, %1, off"
                             :: "v"(lds), "v"(gp) : "memory");
            }
        }

        // ---- gx tiles for this chunk: exact fp32 WMMA (memory-bound phase) ----
        const float* xb = &xc[cur][0];
        v8f gxr[TT];
#pragma unroll
        for (int s = 0; s < TT; ++s) gxr[s] = vzero;
#pragma unroll
        for (int ks = 0; ks < 16; ++ks) {
            int c = 4 * ks + 2 * half;
#pragma unroll
            for (int s = 0; s < TT; ++s) {     // 8 independent acc chains -> ILP
                v2f a;
                a.x = xb[s * XPLANE + nl * XROW + c];
                a.y = xb[s * XPLANE + nl * XROW + c + 1];
                gxr[s] = __builtin_amdgcn_wmma_f32_16x16x4_f32(
                    false, a, false, wih[ks], (short)0, gxr[s], false, false);
            }
        }

        // ---- recurrence: one bf16 WMMA per step (K = H = 32 fits exactly) ----
#pragma unroll
        for (int s = 0; s < TT; ++s) {
            // A = h_prev (16x32 bf16): two aligned b128 LDS loads per lane.
            v8bf alo = *(const v8bf*)(&hb[nl][16 * half]);
            v8bf ahi = *(const v8bf*)(&hb[nl][16 * half + 8]);
            v16bf a = __builtin_shufflevector(alo, ahi,
                                              0, 1, 2, 3, 4, 5, 6, 7,
                                              8, 9, 10, 11, 12, 13, 14, 15);
            v8f gh = __builtin_amdgcn_wmma_f32_16x16x32_bf16(
                false, a, false, whh, (short)0, vzero, false, false);

            if (wv < 4) {
                // r (waves 0,1) / z (waves 2,3): sigmoid(gx + b_ih + gh + b_hh)
                float (*buf)[H_DIM + 1] = (wv < 2) ? rbuf : zbuf;
                int col = ((wv & 1) ? 16 : 0) + nl;
#pragma unroll
                for (int j = 0; j < 8; ++j) {
                    float sg = gxr[s][j] + gh[j] + bias_rz;
                    buf[j + 8 * half][col] = fast_sigmoid(sg);
                }
                __syncthreads();   // gates published
                __syncthreads();   // h updated by waves 4,5
            } else {
                // n gate + state update (waves 4,5)
                float xn[8], hn[8];
#pragma unroll
                for (int j = 0; j < 8; ++j) {
                    xn[j] = gxr[s][j] + bih;      // b_hh must NOT enter xn
                    hn[j] = gh[j] + bhh;          // r multiplies (gh + b_hh) only
                }
                __syncthreads();   // wait for r, z
                int col = ((wv & 1) ? 16 : 0) + nl;
#pragma unroll
                for (int j = 0; j < 8; ++j) {
                    int m    = j + 8 * half;
                    float r  = rbuf[m][col];
                    float z  = zbuf[m][col];
                    float hp = hf[m][col];        // f32 carry path
                    float n  = fast_tanh(xn[j] + r * hn[j]);
                    float hnew = (1.f - z) * n + z * hp;
                    hf[m][col] = hnew;
                    hb[m][col] = (__bf16)hnew;    // bf16 copy feeds next WMMA
                }
                __syncthreads();   // h visible to all waves
            }
        }

        // next chunk's async data must be in LDS before anyone reads it
        asm volatile("s_wait_asynccnt 0x0" ::: "memory");
        __syncthreads();
        cur ^= 1;
    }

    // ---- final hidden state ----
    for (int i = tid; i < BT * H_DIM; i += blockDim.x) {
        int m = i >> 5, k = i & 31;
        out[(size_t)(b0 + m) * H_DIM + k] = hf[m][k];
    }
}

extern "C" void kernel_launch(void* const* d_in, const int* in_sizes, int n_in,
                              void* d_out, int out_size, void* d_ws, size_t ws_size,
                              hipStream_t stream) {
    const float* x    = (const float*)d_in[0];
    const float* W_ih = (const float*)d_in[1];
    const float* W_hh = (const float*)d_in[2];
    const float* b_ih = (const float*)d_in[3];
    const float* b_hh = (const float*)d_in[4];
    float* out = (float*)d_out;

    gru_fused_kernel<<<B_TOT / BT, 192, 0, stream>>>(x, W_ih, W_hh, b_ih, b_hh, out);
}
